// GPT_23106924053321
// MI455X (gfx1250) — matrix operand order repaired and checked by reading, not compiled, not run on hardware
//
#include <hip/hip_runtime.h>
#include <hip/hip_bf16.h>

// ---------------------------------------------------------------------------
// GPT-2 forward for gfx1250 (MI455X).  All GEMMs + attention matmuls go
// through v_wmma_f32_16x16x32_f16 (f16 operands, f32 accumulate).
// GEMM: 128x128 block tile, 8 waves x (32x64) sub-tiles, double-buffered LDS,
// one barrier per K-step, global loads software-pipelined over the WMMAs.
// ---------------------------------------------------------------------------

typedef _Float16 v16h __attribute__((ext_vector_type(16)));
typedef _Float16 v8h  __attribute__((ext_vector_type(8)));
typedef float    v8f  __attribute__((ext_vector_type(8)));

__device__ __forceinline__ v8f wmma_f16(v16h a, v16h b, v8f c) {
  // (neg_a, A, neg_b, B, c_mod, C, reuse_a, reuse_b)
  return __builtin_amdgcn_wmma_f32_16x16x32_f16(false, a, false, b,
                                                (short)0, c, false, false);
}

// Build a 16-half A-fragment from two contiguous 8-half chunks in LDS.
// Contiguous k order used here (B reads match): lane L holds K = 16*(L/16)+{0..7} in v0-3 and
// K = 16*(L/16)+8+{0..7} in v4-7 for row M = L%16.
__device__ __forceinline__ v16h load_a_frag(const _Float16* row_base) {
  v8h lo = *(const v8h*)(row_base);
  v8h hi = *(const v8h*)(row_base + 8);
  v16h r;
#pragma unroll
  for (int i = 0; i < 8; ++i) { r[i] = lo[i]; r[i + 8] = hi[i]; }
  return r;
}

// ---------------------------------------------------------------------------
// Embedding: x[b,s,:] = wte[idx[b,s],:] + wpe[s,:]
// ---------------------------------------------------------------------------
__global__ __launch_bounds__(256) void embed_kernel(
    const int* __restrict__ idx, const float* __restrict__ wte,
    const float* __restrict__ wpe, float* __restrict__ x) {
  const int E = 1024, S = 1024;
  const int row = blockIdx.x;           // b*S + s
  const int s   = row & (S - 1);
  const int tok = idx[row];
#pragma unroll
  for (int i = 0; i < 4; ++i) {
    int e = threadIdx.x + i * 256;
    x[(size_t)row * E + e] = wte[(size_t)tok * E + e] + wpe[(size_t)s * E + e];
  }
}

// ---------------------------------------------------------------------------
// LayerNorm over E=1024: one row per workgroup (256 threads = 8 waves).
// ---------------------------------------------------------------------------
__global__ __launch_bounds__(256) void ln_kernel(
    const float* __restrict__ x, const float* __restrict__ w,
    const float* __restrict__ bia, float* __restrict__ y) {
  const int E = 1024;
  const int row = blockIdx.x;
  const int tid = threadIdx.x;
  const float* xr = x + (size_t)row * E;
  float v[4], s = 0.f, s2 = 0.f;
#pragma unroll
  for (int i = 0; i < 4; ++i) {
    float t = xr[tid + i * 256];
    v[i] = t; s += t; s2 += t * t;
  }
#pragma unroll
  for (int o = 1; o < 32; o <<= 1) {
    s  += __shfl_xor(s,  o, 32);
    s2 += __shfl_xor(s2, o, 32);
  }
  __shared__ float red[2][8];
  const int wave = tid >> 5, lane = tid & 31;
  if (lane == 0) { red[0][wave] = s; red[1][wave] = s2; }
  __syncthreads();
  s = 0.f; s2 = 0.f;
#pragma unroll
  for (int i = 0; i < 8; ++i) { s += red[0][i]; s2 += red[1][i]; }
  const float mean = s * (1.f / E);
  const float var  = s2 * (1.f / E) - mean * mean;
  const float rstd = rsqrtf(var + 1e-5f);
  float* yr = y + (size_t)row * E;
#pragma unroll
  for (int i = 0; i < 4; ++i) {
    int c = tid + i * 256;
    yr[c] = (v[i] - mean) * rstd * w[c] + bia[c];
  }
}

// ---------------------------------------------------------------------------
// WMMA GEMM helpers: stage a 128x32 A-slab and 128x32 (transposed) B-slab.
// A tile f32->f16 row-major [m][k]; B tile transposed [n][k] so the WMMA
// B-fragment (lane n = L%16, 16 contiguous k at 16*(L/16)) is one v16h read.
// ---------------------------------------------------------------------------
__device__ __forceinline__ void gemm_stage_load(
    const float* __restrict__ A, const float* __restrict__ Bw,
    int m0, int n0, int N, int K, int k0, int kpre, int tid,
    float4* fa, float4* fb) {
#pragma unroll
  for (int p = 0; p < 4; ++p) {
    int e = (p * 256 + tid) * 4;        // 0..4092
    int m = e >> 5, k = e & 31;         // A: m 0..127, k 0..28 (x4)
    const float* ap = A + (size_t)(m0 + m) * K + k0 + k;
    __builtin_prefetch(A + (size_t)(m0 + m) * K + kpre + k, 0, 1);
    fa[p] = *(const float4*)ap;
    int gn = n0 + m;                    // B: n 0..127 (same decomposition)
    if (gn < N) {
      const float* bp = Bw + (size_t)(k0 + k) * N + gn;
      fb[p] = make_float4(bp[0], bp[(size_t)N], bp[2 * (size_t)N],
                          bp[3 * (size_t)N]);
    } else {
      fb[p] = make_float4(0.f, 0.f, 0.f, 0.f);
    }
  }
}

__device__ __forceinline__ void gemm_stage_store(
    _Float16 (*__restrict__ As)[32], _Float16 (*__restrict__ Bs)[32],
    int tid, const float4* fa, const float4* fb) {
#pragma unroll
  for (int p = 0; p < 4; ++p) {
    int e = (p * 256 + tid) * 4;
    int m = e >> 5, k = e & 31;
    As[m][k]     = (_Float16)fa[p].x;
    As[m][k + 1] = (_Float16)fa[p].y;
    As[m][k + 2] = (_Float16)fa[p].z;
    As[m][k + 3] = (_Float16)fa[p].w;
    Bs[m][k]     = (_Float16)fb[p].x;
    Bs[m][k + 1] = (_Float16)fb[p].y;
    Bs[m][k + 2] = (_Float16)fb[p].z;
    Bs[m][k + 3] = (_Float16)fb[p].w;
  }
}

// ---------------------------------------------------------------------------
// WMMA GEMM: C[M,N] = act(A[M,K] @ B[K,N] + bias) (+ resid)
// Block = 256 threads (8 waves), tile 128x128, K-step 32, double-buffered.
// Wave (4x2 arrangement) owns a 32x64 sub-tile: 2 A-frags x 4 B-frags =
// 8 v_wmma per K-step; one barrier per K-step.
// ---------------------------------------------------------------------------
__global__ __launch_bounds__(256) void gemm_kernel(
    const float* __restrict__ A, const float* __restrict__ Bw,
    const float* __restrict__ bias, const float* __restrict__ resid,
    float* __restrict__ C, int M, int N, int K, int gelu) {
  __shared__ __attribute__((aligned(32))) _Float16 As[2][128][32];
  __shared__ __attribute__((aligned(32))) _Float16 Bs[2][128][32];  // [n][k]

  const int tid  = threadIdx.x;
  const int lane = tid & 31;
  const int wave = tid >> 5;
  const int hi   = lane >> 4;         // half-wave id
  const int l16  = lane & 15;
  const int n0   = blockIdx.x * 128;
  const int m0   = blockIdx.y * 128;
  const int wm   = (wave >> 1) * 32;  // wave sub-tile origin (rows)
  const int wn   = (wave & 1) * 64;   // wave sub-tile origin (cols)

  v8f acc[2][4] = {};
  float4 fa[4], fb[4];

  // ---- prologue: stage K-slab 0 into buffer 0 ----
  {
    int kpre = (32 < K) ? 32 : 0;
    gemm_stage_load(A, Bw, m0, n0, N, K, 0, kpre, tid, fa, fb);
    gemm_stage_store(As[0], Bs[0], tid, fa, fb);
  }
  __syncthreads();

  int buf = 0;
  for (int k0 = 0; k0 < K; k0 += 32, buf ^= 1) {
    const bool has_next = (k0 + 32) < K;
    // issue global loads for the next slab first (latency hidden by WMMAs)
    if (has_next) {
      int kpre = (k0 + 64 < K) ? (k0 + 64) : (K - 32);
      gemm_stage_load(A, Bw, m0, n0, N, K, k0 + 32, kpre, tid, fa, fb);
    }
    // ---- compute on current buffer: 2 x 4 WMMAs ----
    v16h af[2], bfv[4];
#pragma unroll
    for (int i = 0; i < 2; ++i)
      af[i] = load_a_frag(&As[buf][wm + i * 16 + l16][hi * 16]);
#pragma unroll
    for (int j = 0; j < 4; ++j)
      bfv[j] = *(const v16h*)&Bs[buf][wn + j * 16 + l16][hi * 16];
#pragma unroll
    for (int i = 0; i < 2; ++i)
#pragma unroll
      for (int j = 0; j < 4; ++j)
        acc[i][j] = wmma_f16(af[i], bfv[j], acc[i][j]);
    // ---- publish next slab; single barrier both publishes and protects ----
    if (has_next)
      gemm_stage_store(As[buf ^ 1], Bs[buf ^ 1], tid, fa, fb);
    __syncthreads();
  }

  // ---- epilogue: bias (+gelu) (+resid); C layout: row = r + 8*hi, col = l16
#pragma unroll
  for (int i = 0; i < 2; ++i) {
#pragma unroll
    for (int j = 0; j < 4; ++j) {
      int col = n0 + wn + j * 16 + l16;
      if (col < N) {
        float bv = bias[col];
#pragma unroll
        for (int r = 0; r < 8; ++r) {
          int row = m0 + wm + i * 16 + r + 8 * hi;
          float v = acc[i][j][r] + bv;
          if (gelu) v = 0.5f * v * (1.f + erff(v * 0.70710678118654752f));
          if (resid) v += resid[(size_t)row * N + col];
          C[(size_t)row * N + col] = v;
        }
      }
    }
  }
}

// ---------------------------------------------------------------------------
// Flash attention: grid (S/64, H, B), block 128 (4 waves x 16 query rows).
// qkv layout: [B*S][3E], q at 0, k at E, v at 2E; head h at h*D (D=64).
// Q.K^T and P.V both via WMMA; online softmax with half-wave shfl reductions.
// ---------------------------------------------------------------------------
__global__ __launch_bounds__(128) void attn_kernel(
    const float* __restrict__ qkv, float* __restrict__ y) {
  const int S = 1024, E = 1024, D = 64;
  const int qt = blockIdx.x, h = blockIdx.y, b = blockIdx.z;
  const int tid = threadIdx.x, lane = tid & 31, wave = tid >> 5;
  const int hi = lane >> 4, l16 = lane & 15;
  const size_t row3E = (size_t)3 * E;

  __shared__ __attribute__((aligned(32))) _Float16 Kt[64][64];  // [krow][d]
  __shared__ __attribute__((aligned(32))) _Float16 Vt[64][64];  // [d][krow]
  __shared__ __attribute__((aligned(32))) _Float16 Pt[64][64];  // probs, wave rows

  // ---- Q fragments (A layout), loaded once straight from global ----
  const int qrow = qt * 64 + wave * 16 + l16;
  const float* qp = qkv + (size_t)(b * S + qrow) * row3E + h * D;
  v16h qf[2];
#pragma unroll
  for (int c = 0; c < 2; ++c) {
    int base = c * 32 + hi * 16;
    float4 f0 = *(const float4*)(qp + base);
    float4 f1 = *(const float4*)(qp + base + 4);
    float4 f2 = *(const float4*)(qp + base + 8);
    float4 f3 = *(const float4*)(qp + base + 12);
    v16h t;
    t[0] = (_Float16)f0.x; t[1] = (_Float16)f0.y; t[2] = (_Float16)f0.z; t[3] = (_Float16)f0.w;
    t[4] = (_Float16)f1.x; t[5] = (_Float16)f1.y; t[6] = (_Float16)f1.z; t[7] = (_Float16)f1.w;
    t[8]  = (_Float16)f2.x; t[9]  = (_Float16)f2.y; t[10] = (_Float16)f2.z; t[11] = (_Float16)f2.w;
    t[12] = (_Float16)f3.x; t[13] = (_Float16)f3.y; t[14] = (_Float16)f3.z; t[15] = (_Float16)f3.w;
    qf[c] = t;
  }

  v8f oacc[4] = {};
  float mrow[8], lrow[8];
#pragma unroll
  for (int r = 0; r < 8; ++r) { mrow[r] = -3.0e38f; lrow[r] = 0.f; }

  for (int kt = 0; kt <= qt; ++kt) {
    const int kbase = kt * 64;
    // ---- stage K (row-major) and V (transposed) tiles as f16 ----
#pragma unroll
    for (int p = 0; p < 8; ++p) {
      int e = (p * 128 + tid) * 4;  // 0..4092
      {
        int r = e >> 6, d = e & 63;
        const float4 f = *(const float4*)(qkv +
            (size_t)(b * S + kbase + r) * row3E + E + h * D + d);
        Kt[r][d]     = (_Float16)f.x;
        Kt[r][d + 1] = (_Float16)f.y;
        Kt[r][d + 2] = (_Float16)f.z;
        Kt[r][d + 3] = (_Float16)f.w;
      }
      {
        int d = e >> 6, r0 = e & 63;
        const float* vp = qkv + (size_t)(b * S + kbase + r0) * row3E + 2 * E + h * D + d;
        Vt[d][r0]     = (_Float16)vp[0];
        Vt[d][r0 + 1] = (_Float16)vp[row3E];
        Vt[d][r0 + 2] = (_Float16)vp[2 * row3E];
        Vt[d][r0 + 3] = (_Float16)vp[3 * row3E];
      }
    }
    __syncthreads();

    // ---- S = Q K^T (16x64 per wave): 4 n-tiles x 2 d-chunks ----
    v8f sa[4] = {};
#pragma unroll
    for (int ni = 0; ni < 4; ++ni) {
#pragma unroll
      for (int c = 0; c < 2; ++c) {
        v16h bfrag = *(const v16h*)&Kt[ni * 16 + l16][c * 32 + hi * 16];
        sa[ni] = wmma_f16(qf[c], bfrag, sa[ni]);
      }
    }

    // ---- scale + causal mask ----
    const int rowg0 = qt * 64 + wave * 16 + 8 * hi;
#pragma unroll
    for (int ni = 0; ni < 4; ++ni) {
      int colg = kbase + ni * 16 + l16;
#pragma unroll
      for (int r = 0; r < 8; ++r) {
        float v = sa[ni][r] * 0.125f;          // 1/sqrt(64)
        if (colg > rowg0 + r) v = -1e9f;
        sa[ni][r] = v;
      }
    }

    // ---- online softmax (rows live across 16 lanes of a half-wave) ----
#pragma unroll
    for (int r = 0; r < 8; ++r) {
      float mloc = fmaxf(fmaxf(sa[0][r], sa[1][r]), fmaxf(sa[2][r], sa[3][r]));
#pragma unroll
      for (int o = 1; o < 16; o <<= 1) mloc = fmaxf(mloc, __shfl_xor(mloc, o, 32));
      float mnew  = fmaxf(mrow[r], mloc);
      float alpha = __expf(mrow[r] - mnew);
      float lsum  = 0.f;
#pragma unroll
      for (int ni = 0; ni < 4; ++ni) {
        float pv = __expf(sa[ni][r] - mnew);
        sa[ni][r] = pv;
        lsum += pv;
      }
#pragma unroll
      for (int o = 1; o < 16; o <<= 1) lsum += __shfl_xor(lsum, o, 32);
      lrow[r] = lrow[r] * alpha + lsum;
      mrow[r] = mnew;
#pragma unroll
      for (int dt = 0; dt < 4; ++dt) oacc[dt][r] *= alpha;
    }

    // ---- round-trip P through LDS to reshape C-layout -> A-fragment ----
#pragma unroll
    for (int ni = 0; ni < 4; ++ni)
#pragma unroll
      for (int r = 0; r < 8; ++r)
        Pt[wave * 16 + r + 8 * hi][ni * 16 + l16] = (_Float16)sa[ni][r];

    v16h pf[2];
#pragma unroll
    for (int c = 0; c < 2; ++c)
      pf[c] = load_a_frag(&Pt[wave * 16 + l16][c * 32 + hi * 16]);

    // ---- O += P V : 4 d-tiles x 2 k-chunks ----
#pragma unroll
    for (int dt = 0; dt < 4; ++dt)
#pragma unroll
      for (int c = 0; c < 2; ++c) {
        v16h bfrag = *(const v16h*)&Vt[dt * 16 + l16][c * 32 + hi * 16];
        oacc[dt] = wmma_f16(pf[c], bfrag, oacc[dt]);
      }
    __syncthreads();
  }

  // ---- O / l, store to y[b, q, h, d] (== [B*S, E] with h*D + d) ----
#pragma unroll
  for (int dt = 0; dt < 4; ++dt) {
    int d = h * D + dt * 16 + l16;
#pragma unroll
    for (int r = 0; r < 8; ++r) {
      int rowg = qt * 64 + wave * 16 + r + 8 * hi;
      y[(size_t)(b * S + rowg) * E + d] = oacc[dt][r] / lrow[r];
    }
  }
}

// ---------------------------------------------------------------------------
// Host-side orchestration (graph-capture safe: only kernel launches on stream)
// ---------------------------------------------------------------------------
extern "C" void kernel_launch(void* const* d_in, const int* in_sizes, int n_in,
                              void* d_out, int out_size, void* d_ws, size_t ws_size,
                              hipStream_t stream) {
  (void)in_sizes; (void)n_in; (void)out_size; (void)ws_size;

  const int*   idx    = (const int*)  d_in[0];
  const float* wte    = (const float*)d_in[1];
  const float* wpe    = (const float*)d_in[2];
  const float* ln1_w  = (const float*)d_in[3];
  const float* ln1_b  = (const float*)d_in[4];
  const float* qkv_w  = (const float*)d_in[5];
  const float* qkv_b  = (const float*)d_in[6];
  const float* out_w  = (const float*)d_in[7];
  const float* out_b  = (const float*)d_in[8];
  const float* ln2_w  = (const float*)d_in[9];
  const float* ln2_b  = (const float*)d_in[10];
  const float* fc1_w  = (const float*)d_in[11];
  const float* fc1_b  = (const float*)d_in[12];
  const float* fc2_w  = (const float*)d_in[13];
  const float* fc2_b  = (const float*)d_in[14];
  const float* lnf_w  = (const float*)d_in[15];
  const float* lnf_b  = (const float*)d_in[16];
  const float* head_w = (const float*)d_in[17];
  const float* head_b = (const float*)d_in[18];
  float* out = (float*)d_out;

  const int Bn = 4, S = 1024, E = 1024, Hh = 16, HID = 4096, L = 8, V = 1000;
  const int M = Bn * S;  // 4096 rows

  // Workspace: x | h | y | big   (big holds qkv [M,3E] or MLP hidden [M,HID])
  float* x   = (float*)d_ws;
  float* h   = x + (size_t)M * E;
  float* y   = h + (size_t)M * E;
  float* big = y + (size_t)M * E;

  embed_kernel<<<M, 256, 0, stream>>>(idx, wte, wpe, x);

  for (int l = 0; l < L; ++l) {
    // attention block
    ln_kernel<<<M, 256, 0, stream>>>(x, ln1_w + l * E, ln1_b + l * E, h);
    gemm_kernel<<<dim3(3 * E / 128, M / 128), 256, 0, stream>>>(
        h, qkv_w + (size_t)l * E * 3 * E, qkv_b + (size_t)l * 3 * E,
        nullptr, big, M, 3 * E, E, 0);
    attn_kernel<<<dim3(S / 64, Hh, Bn), 128, 0, stream>>>(big, y);
    gemm_kernel<<<dim3(E / 128, M / 128), 256, 0, stream>>>(
        y, out_w + (size_t)l * E * E, out_b + (size_t)l * E,
        x, x, M, E, E, 0);
    // MLP block
    ln_kernel<<<M, 256, 0, stream>>>(x, ln2_w + l * E, ln2_b + l * E, h);
    gemm_kernel<<<dim3(HID / 128, M / 128), 256, 0, stream>>>(
        h, fc1_w + (size_t)l * E * HID, fc1_b + (size_t)l * HID,
        nullptr, big, M, HID, E, 1);
    gemm_kernel<<<dim3(E / 128, M / 128), 256, 0, stream>>>(
        big, fc2_w + (size_t)l * HID * E, fc2_b + (size_t)l * E,
        x, x, M, E, HID, 0);
  }

  ln_kernel<<<M, 256, 0, stream>>>(x, lnf_w, lnf_b, h);
  gemm_kernel<<<dim3((V + 127) / 128, M / 128), 256, 0, stream>>>(
      h, head_w, head_b, nullptr, out, M, V, E, 0);
}
